// LinAngularAttention_ksh_3435973836892
// MI455X (gfx1250) — matrix operand order repaired and checked by reading, not compiled, not run on hardware
//
#include <hip/hip_runtime.h>
#include <hip/hip_bf16.h>

// LinAngularAttention for MI455X (gfx1250), wave32, bf16 WMMA pipeline.
// Shapes fixed by reference: B=8, N=4096, C=384, nh=8, D=F=48.

#define B_SZ   8
#define N_TOK  4096
#define C_DIM  384
#define NH     8
#define HD     48
#define M_TOT  (B_SZ * N_TOK)      // 32768 token rows

typedef __attribute__((ext_vector_type(16))) __bf16 bf16x16;
typedef __attribute__((ext_vector_type(8)))  float  f32x8;

union Frag32B {                    // one WMMA 16-bit operand fragment per lane
  uint4          q[2];
  unsigned short s[16];
  bf16x16        v;
};

__device__ __forceinline__ unsigned short f2bf(float x) {
  union { float f; unsigned u; } t; t.f = x;
  unsigned r = t.u + 0x7FFFu + ((t.u >> 16) & 1u);   // round-to-nearest-even
  return (unsigned short)(r >> 16);
}

__device__ __forceinline__ f32x8 wmma_bf16(bf16x16 a, bf16x16 b, f32x8 c) {
  // D = A(16x32 bf16) x B(32x16 bf16) + C(16x16 f32)
  return __builtin_amdgcn_wmma_f32_16x16x32_bf16(false, a, false, b, (short)0, c,
                                                 false, false);
}

__device__ __forceinline__ f32x8 zero8() {
  f32x8 z = {0.f, 0.f, 0.f, 0.f, 0.f, 0.f, 0.f, 0.f};
  return z;
}

// CDNA5 async memory->LDS copy (VGLOBAL GLOBAL_LOAD_ASYNC_TO_LDS_B128,
// tracked by ASYNCcnt). 16 bytes per lane, no VGPR staging.
__device__ __forceinline__ void async_ld16(void* lds_dst, const void* gsrc) {
  unsigned l = (unsigned)(size_t)lds_dst;            // LDS byte address (AS 3)
  asm volatile("global_load_async_to_lds_b128 %0, %1, off"
               :: "v"(l), "v"(gsrc)
               : "memory");
}
__device__ __forceinline__ void wait_async0() {
  asm volatile("s_wait_asynccnt 0" ::: "memory");
}

// ---------------------------------------------------------------------------
// Kernel 0: fp32 -> bf16 conversions + zero the attn accumulator (every call,
// since the harness does not re-zero the workspace between replays).
// ---------------------------------------------------------------------------
__global__ void prep_kernel(const float* __restrict__ x,
                            const float* __restrict__ w_v,
                            const float* __restrict__ w_proj,
                            unsigned short* __restrict__ x_bf,
                            unsigned short* __restrict__ wv_bf,
                            unsigned short* __restrict__ wp_bf,
                            float* __restrict__ attnT) {
  size_t i = (size_t)blockIdx.x * blockDim.x + threadIdx.x;
  if (i < (size_t)M_TOT * C_DIM) x_bf[i] = f2bf(x[i]);
  if (i < (size_t)C_DIM * C_DIM) {
    wv_bf[i] = f2bf(w_v[i]);
    wp_bf[i] = f2bf(w_proj[i]);
  }
  if (i < (size_t)B_SZ * NH * HD * 64) attnT[i] = 0.f;   // padded [bh][48][64]
}

// ---------------------------------------------------------------------------
// Kernel 1: fold hash projection into qk weight.
// W2[h*48+f][c] = sum_d hash[h][f][d] * w_qk[h*48+d][c]   (sign() is scale
// invariant, so the qk L2-normalize is dropped entirely).
// ---------------------------------------------------------------------------
__global__ void combine_kernel(const float* __restrict__ hash,
                               const float* __restrict__ w_qk,
                               unsigned short* __restrict__ w2_bf) {
  int i = blockIdx.x * blockDim.x + threadIdx.x;       // < 384*384
  int row = i / C_DIM;
  int c   = i % C_DIM;
  int h = row / HD, f = row % HD;
  const float* hp = hash + ((size_t)h * HD + f) * HD;
  float acc = 0.f;
  for (int d = 0; d < HD; ++d)
    acc += hp[d] * w_qk[((size_t)h * HD + d) * C_DIM + c];
  w2_bf[i] = f2bf(acc);
}

// ---------------------------------------------------------------------------
// Fused projection GEMM: one pass over x_bf produces BOTH
//   codes = sign(x @ W2^T)/sqrt(48)   (bf16, row-major + per-head transposed)
//   v     = x @ w_v^T                 (f32 row-major + per-head transposed bf16)
// 512 threads = 16 waves. A tile (128x64) shared in LDS; waves 0-7 multiply
// against the W2 tile, waves 8-15 against the w_v tile. Each wave owns a
// 64x32 sub-tile = 4x2 WMMA fragments. LDS filled with async-to-LDS copies.
// ---------------------------------------------------------------------------
__global__ __launch_bounds__(512)
void gemm_qkv_fused_kernel(const unsigned short* __restrict__ x_bf,
                           const unsigned short* __restrict__ w2_bf,
                           const unsigned short* __restrict__ wv_bf,
                           unsigned short* __restrict__ codes,
                           unsigned short* __restrict__ codesT,
                           float* __restrict__ v_f32,
                           unsigned short* __restrict__ vT) {
  __shared__ uint4 lA[128 * 9];     // 128 rows x (64 bf16 + 8 pad) = 9 uint4/row
  __shared__ uint4 lB2[128 * 9];
  __shared__ uint4 lBv[128 * 9];

  const int tid  = threadIdx.x;
  const int lane = tid & 31;
  const int wv   = tid >> 5;        // 0..15
  const int hf   = lane >> 4;
  const int l16  = lane & 15;
  const int mBase = blockIdx.x * 128;
  const int nBase = blockIdx.y * 128;
  const int outsel = wv >> 3;       // 0 -> codes, 1 -> v
  const int sub = wv & 7;
  const int wm = (sub >> 2) * 64;
  const int wn = (sub & 3) * 32;
  const uint4* lB = outsel ? lBv : lB2;

  f32x8 acc[4][2];
  for (int i = 0; i < 4; ++i)
    for (int j = 0; j < 2; ++j) acc[i][j] = zero8();

  for (int k0 = 0; k0 < C_DIM; k0 += 64) {
    // async fill: 3 tiles x 1024 uint4, 512 threads -> 6 copies per thread
    for (int t = 0; t < 2; ++t) {
      int u = tid + t * 512;                 // 0..1023
      int row = u >> 3, c16 = u & 7;
      size_t ga = (size_t)(mBase + row) * C_DIM + k0 + c16 * 8;
      size_t gb = (size_t)(nBase + row) * C_DIM + k0 + c16 * 8;
      async_ld16(&lA[row * 9 + c16],  x_bf  + ga);
      async_ld16(&lB2[row * 9 + c16], w2_bf + gb);
      async_ld16(&lBv[row * 9 + c16], wv_bf + gb);
    }
    wait_async0();
    __syncthreads();

    for (int ks = 0; ks < 64; ks += 32) {
      Frag32B fa[4], fb[2];
      for (int i = 0; i < 4; ++i) {          // A: lane = M row, two 16B k-runs
        int r  = wm + i * 16 + l16;
        int kf = ks + hf * 8;
        fa[i].q[0] = lA[(r * 72 + kf) >> 3];
        fa[i].q[1] = lA[(r * 72 + kf + 16) >> 3];
      }
      for (int j = 0; j < 2; ++j) {          // B: lane = N col, two 16B k-runs
        int r  = wn + j * 16 + l16;
        int kf = ks + hf * 8;
        int qi = (r * 72 + kf) >> 3;
        fb[j].q[0] = lB[qi];
        fb[j].q[1] = lB[qi + 2];
      }
      for (int i = 0; i < 4; ++i)
        for (int j = 0; j < 2; ++j)
          acc[i][j] = wmma_bf16(fa[i].v, fb[j].v, acc[i][j]);
    }
    __syncthreads();
  }

  // epilogue: D frag -> lane holds col = l16, VGPR p -> row = hf*8+p
  for (int i = 0; i < 4; ++i)
    for (int j = 0; j < 2; ++j)
      for (int p = 0; p < 8; ++p) {
        int row = mBase + wm + i * 16 + hf * 8 + p;     // token
        int col = nBase + wn + j * 16 + l16;            // channel
        int h = col / HD, d = col % HD;
        int b = row >> 12, nt = row & (N_TOK - 1);
        float a = acc[i][j][p];
        if (outsel == 0) {
          float s = (a > 0.f ? 1.f : (a < 0.f ? -1.f : 0.f)) * 0.14433756729740643f;
          unsigned short bv = f2bf(s);
          codes[(size_t)row * C_DIM + col] = bv;
          codesT[((size_t)(b * NH + h) * HD + d) * N_TOK + nt] = bv;
        } else {
          v_f32[(size_t)row * C_DIM + col] = a;
          vT[((size_t)(b * NH + h) * HD + d) * N_TOK + nt] = f2bf(a);
        }
      }
}

// ---------------------------------------------------------------------------
// Final projection GEMM: out = u @ w_proj^T + b_proj (f32 output).
// 256 threads = 8 waves, 128x128 tile, async LDS fill.
// ---------------------------------------------------------------------------
__global__ __launch_bounds__(256)
void gemm_proj_kernel(const unsigned short* __restrict__ Abuf,
                      const unsigned short* __restrict__ Wbuf,
                      float* __restrict__ out_f32,
                      const float* __restrict__ bias) {
  __shared__ uint4 lA[128 * 9];
  __shared__ uint4 lB[128 * 9];

  const int tid  = threadIdx.x;
  const int lane = tid & 31;
  const int wv   = tid >> 5;
  const int hf   = lane >> 4;
  const int l16  = lane & 15;
  const int mBase = blockIdx.x * 128;
  const int nBase = blockIdx.y * 128;
  const int wm = (wv >> 2) * 64;
  const int wn = (wv & 3) * 32;

  f32x8 acc[4][2];
  for (int i = 0; i < 4; ++i)
    for (int j = 0; j < 2; ++j) acc[i][j] = zero8();

  for (int k0 = 0; k0 < C_DIM; k0 += 64) {
    for (int t = 0; t < 4; ++t) {
      int u = tid + t * 256;                 // 0..1023
      int row = u >> 3, c16 = u & 7;
      async_ld16(&lA[row * 9 + c16], Abuf + (size_t)(mBase + row) * C_DIM + k0 + c16 * 8);
      async_ld16(&lB[row * 9 + c16], Wbuf + (size_t)(nBase + row) * C_DIM + k0 + c16 * 8);
    }
    wait_async0();
    __syncthreads();

    for (int ks = 0; ks < 64; ks += 32) {
      Frag32B fa[4], fb[2];
      for (int i = 0; i < 4; ++i) {
        int r  = wm + i * 16 + l16;
        int kf = ks + hf * 8;
        fa[i].q[0] = lA[(r * 72 + kf) >> 3];
        fa[i].q[1] = lA[(r * 72 + kf + 16) >> 3];
      }
      for (int j = 0; j < 2; ++j) {
        int r  = wn + j * 16 + l16;
        int kf = ks + hf * 8;
        int qi = (r * 72 + kf) >> 3;
        fb[j].q[0] = lB[qi];
        fb[j].q[1] = lB[qi + 2];
      }
      for (int i = 0; i < 4; ++i)
        for (int j = 0; j < 2; ++j)
          acc[i][j] = wmma_bf16(fa[i].v, fb[j].v, acc[i][j]);
    }
    __syncthreads();
  }

  for (int i = 0; i < 4; ++i)
    for (int j = 0; j < 2; ++j)
      for (int p = 0; p < 8; ++p) {
        int row = mBase + wm + i * 16 + hf * 8 + p;
        int col = nBase + wn + j * 16 + l16;
        out_f32[(size_t)row * C_DIM + col] = acc[i][j][p] + bias[col];
      }
}

// ---------------------------------------------------------------------------
// Stage 2: attn^T[d][f] = sum_n codes[n,f] * v[n,d] per (b,h).
// A = codesT (rows f, K=token contiguous), B = vT (rows d, K contiguous).
// K=4096 split 8 ways, f32 atomics into padded attnT[bh][48][64].
// ---------------------------------------------------------------------------
__global__ __launch_bounds__(32)
void attn_kernel(const unsigned short* __restrict__ codesT,
                 const unsigned short* __restrict__ vT,
                 float* __restrict__ attnT) {
  const int lane = threadIdx.x;
  const int hf = lane >> 4, l16 = lane & 15;
  const int bh = blockIdx.x >> 3;
  const int ck = blockIdx.x & 7;
  const unsigned short* cb = codesT + (size_t)bh * HD * N_TOK;
  const unsigned short* vb = vT     + (size_t)bh * HD * N_TOK;

  f32x8 acc[3][3];
  for (int m = 0; m < 3; ++m)
    for (int d = 0; d < 3; ++d) acc[m][d] = zero8();

  const int k0 = ck * 512;
  for (int kk = 0; kk < 512; kk += 32) {
    int k = k0 + kk;
    Frag32B fa[3], fb[3];
    for (int mt = 0; mt < 3; ++mt) {
      const unsigned short* p = cb + (size_t)(mt * 16 + l16) * N_TOK + k;
      fa[mt].q[0] = *(const uint4*)(p + hf * 8);
      fa[mt].q[1] = *(const uint4*)(p + 16 + hf * 8);
    }
    for (int dt = 0; dt < 3; ++dt) {
      const unsigned short* p = vb + (size_t)(dt * 16 + l16) * N_TOK + k + hf * 8;
      fb[dt].q[0] = *(const uint4*)(p);
      fb[dt].q[1] = *(const uint4*)(p + 16);
    }
    for (int mt = 0; mt < 3; ++mt)
      for (int dt = 0; dt < 3; ++dt)
        acc[mt][dt] = wmma_bf16(fa[mt].v, fb[dt].v, acc[mt][dt]);
  }

  float* ab = attnT + (size_t)bh * HD * 64;
  for (int mt = 0; mt < 3; ++mt)
    for (int dt = 0; dt < 3; ++dt)
      for (int p = 0; p < 8; ++p) {
        int f = mt * 16 + hf * 8 + p;
        int d = dt * 16 + l16;
        atomicAdd(&ab[d * 64 + f], acc[mt][dt][p]);
      }
}

// ---------------------------------------------------------------------------
// Stage 3: out = 0.5 v + (1/pi) codes @ attn ; row-L2-normalize over D=48 ;
// + 9-tap depthwise conv along tokens (zero pad); emit bf16 for final GEMM.
// One wave per (b, h, 16-token tile). v tile (24 x 48) staged in LDS w/ halo.
// ---------------------------------------------------------------------------
__global__ __launch_bounds__(32)
void stage3_kernel(const unsigned short* __restrict__ codes_bf,
                   const float* __restrict__ attnT,
                   const float* __restrict__ v_f32,
                   const float* __restrict__ w_dconv,
                   unsigned short* __restrict__ u_bf) {
  __shared__ float vt[24 * 48];
  const int bx = blockIdx.x;
  const int b  = bx >> 11;            // 8*256 tiles per batch image
  const int h  = (bx >> 8) & 7;
  const int nt = bx & 255;
  const int lane = threadIdx.x;
  const int hf = lane >> 4, l16 = lane & 15;

  for (int t = 0; t < 36; ++t) {      // 1152 floats / 32 lanes
    int idx = lane + t * 32;
    int r = idx / 48, d = idx % 48;
    int n = nt * 16 - 4 + r;
    float vv = 0.f;
    if (n >= 0 && n < N_TOK)
      vv = v_f32[(size_t)(b * N_TOK + n) * C_DIM + h * HD + d];
    vt[idx] = vv;
  }
  float wd[9];
  for (int j = 0; j < 9; ++j) wd[j] = w_dconv[h * 9 + j];
  __syncthreads();

  // B fragments: attn^T rows d, K=f padded to 64 with zeros, cvt f32->bf16
  Frag32B fb[3][2];
  const float* ab = attnT + (size_t)(b * NH + h) * HD * 64;
  for (int dt = 0; dt < 3; ++dt)
    for (int kc = 0; kc < 2; ++kc) {
      int d  = dt * 16 + l16;
      int f0 = kc * 32 + hf * 8;
      for (int e = 0; e < 16; ++e)
        fb[dt][kc].s[e] = f2bf(ab[d * 64 + f0 + e + (e >= 8 ? 8 : 0)]);
    }

  // A fragments: codes rows = 16 tokens, K=f (48 real + 16 zero pad)
  const unsigned short* crow =
      codes_bf + (size_t)(b * N_TOK + nt * 16 + l16) * C_DIM + h * HD;
  Frag32B fa[2];
  fa[0].q[0] = *(const uint4*)(crow + hf * 8);
  fa[0].q[1] = *(const uint4*)(crow + 16 + hf * 8);
  fa[1].q[0] = *(const uint4*)(crow + 32 + hf * 8);
  fa[1].q[1] = make_uint4(0u, 0u, 0u, 0u);      // f = 48..63 -> zero

  f32x8 acc[3] = {zero8(), zero8(), zero8()};
  for (int kc = 0; kc < 2; ++kc)
    for (int dt = 0; dt < 3; ++dt)
      acc[dt] = wmma_bf16(fa[kc].v, fb[dt][kc].v, acc[dt]);

  // epilogue: t = 0.5 v + acc/pi ; normalize per token row (48 dims) ; + conv
  float tv[3][8];
  for (int dt = 0; dt < 3; ++dt)
    for (int p = 0; p < 8; ++p) {
      int m = hf * 8 + p;
      int d = dt * 16 + l16;
      tv[dt][p] = 0.5f * vt[(m + 4) * 48 + d] + 0.3183098861837907f * acc[dt][p];
    }
  for (int p = 0; p < 8; ++p) {
    float s = tv[0][p] * tv[0][p] + tv[1][p] * tv[1][p] + tv[2][p] * tv[2][p];
    s += __shfl_xor(s, 1, 32);
    s += __shfl_xor(s, 2, 32);
    s += __shfl_xor(s, 4, 32);
    s += __shfl_xor(s, 8, 32);        // sum over the 16 lanes of this half
    float inv = rsqrtf(s);
    int m = hf * 8 + p;
    for (int dt = 0; dt < 3; ++dt) {
      int d = dt * 16 + l16;
      float dc = 0.f;
      for (int j = 0; j < 9; ++j) dc += wd[j] * vt[(m + j) * 48 + d];
      float u = tv[dt][p] * inv + dc;
      u_bf[(size_t)(b * N_TOK + nt * 16 + m) * C_DIM + h * HD + d] = f2bf(u);
    }
  }
}

// ---------------------------------------------------------------------------
extern "C" void kernel_launch(void* const* d_in, const int* in_sizes, int n_in,
                              void* d_out, int out_size, void* d_ws, size_t ws_size,
                              hipStream_t stream) {
  const float* x       = (const float*)d_in[0];
  const float* w_qk    = (const float*)d_in[1];
  const float* w_v     = (const float*)d_in[2];
  const float* w_proj  = (const float*)d_in[3];
  const float* b_proj  = (const float*)d_in[4];
  const float* hash    = (const float*)d_in[5];
  const float* w_dconv = (const float*)d_in[6];
  float* out = (float*)d_out;

  char* ws = (char*)d_ws;
  size_t off = 0;
  auto carve = [&](size_t bytes) {
    void* p = ws + off;
    off = (off + bytes + 255) & ~(size_t)255;
    return p;
  };
  const size_t TOK_BF = (size_t)M_TOT * C_DIM * sizeof(unsigned short);  // 25.2 MB
  const size_t W_BF   = (size_t)C_DIM * C_DIM * sizeof(unsigned short);

  unsigned short* x_bf    = (unsigned short*)carve(TOK_BF);
  unsigned short* codes   = (unsigned short*)carve(TOK_BF);
  unsigned short* codesT  = (unsigned short*)carve(TOK_BF);
  unsigned short* vT      = (unsigned short*)carve(TOK_BF);
  unsigned short* u_bf    = (unsigned short*)carve(TOK_BF);
  float*          v_f32   = (float*)carve((size_t)M_TOT * C_DIM * sizeof(float));
  unsigned short* w2_bf   = (unsigned short*)carve(W_BF);
  unsigned short* wv_bf   = (unsigned short*)carve(W_BF);
  unsigned short* wp_bf   = (unsigned short*)carve(W_BF);
  float*          attnT   = (float*)carve((size_t)B_SZ * NH * HD * 64 * sizeof(float));
  (void)ws_size; (void)in_sizes; (void)n_in; (void)out_size;

  // 0) conversions + zero attn accumulator
  prep_kernel<<<(M_TOT * C_DIM + 255) / 256, 256, 0, stream>>>(
      x, w_v, w_proj, x_bf, wv_bf, wp_bf, attnT);
  // 1) fold hash into qk weight
  combine_kernel<<<(C_DIM * C_DIM + 255) / 256, 256, 0, stream>>>(hash, w_qk, w2_bf);
  // 2+3) one pass over x: codes (+transposed) AND v (f32 + transposed bf16)
  gemm_qkv_fused_kernel<<<dim3(M_TOT / 128, C_DIM / 128), 512, 0, stream>>>(
      x_bf, w2_bf, wv_bf, codes, codesT, v_f32, vT);
  // 4) attn^T = (codes^T v)^T   per (b,h), K split 8x + atomics
  attn_kernel<<<B_SZ * NH * 8, 32, 0, stream>>>(codesT, vT, attnT);
  // 5) out-core: 0.5v + codes@attn/pi, normalize, +dconv -> bf16
  stage3_kernel<<<B_SZ * NH * (N_TOK / 16), 32, 0, stream>>>(
      codes, attnT, v_f32, w_dconv, u_bf);
  // 6) final projection + bias -> f32 output
  gemm_proj_kernel<<<dim3(M_TOT / 128, C_DIM / 128), 256, 0, stream>>>(
      u_bf, wp_bf, out, b_proj);
}
